// Generator_59605556134524
// MI455X (gfx1250) — matrix operand order repaired and checked by reading, not compiled, not run on hardware
//
#include <hip/hip_runtime.h>
#include <hip/hip_bf16.h>

// ---------------------------------------------------------------------------
// Social-GAN generator forward for MI455X (gfx1250, wave32, WMMA bf16)
// ---------------------------------------------------------------------------
#define OBS_LEN 8
#define PRED_LEN 12
#define NG 32
#define NP 32
#define NB 1024          // NG*NP
#define EMB 64
#define EH 64
#define DH 128
#define BN 1024
#define MLPH 1024

typedef __attribute__((ext_vector_type(16))) __bf16 v16bf;
typedef __attribute__((ext_vector_type(8)))  float  v8f;

union ABFrag { uint4 q[2]; v16bf v; };
union F8     { v8f v; float f[8]; };

__device__ __forceinline__ unsigned short f2bf(float f) {
  unsigned int u = __builtin_bit_cast(unsigned int, f);
  unsigned int r = u + 0x7FFFu + ((u >> 16) & 1u);   // round-to-nearest-even
  return (unsigned short)(r >> 16);
}
__device__ __forceinline__ float sigm(float x) { return 1.f / (1.f + __expf(-x)); }

// A fragment (16x32 bf16, M x K). ISA layout: lane<16 -> M=lane, K in {k0..k0+7, k0+16..k0+23};
// lane>=16 -> M=lane-16, K in {k0+8..k0+15, k0+24..k0+31}.
__device__ __forceinline__ v16bf load_a(const unsigned short* A, int lda, int m0, int k0) {
  int lane = threadIdx.x & 31;
  const unsigned short* p = A + (size_t)(m0 + (lane & 15)) * lda + k0 + ((lane >> 4) << 3);
  ABFrag f;
  f.q[0] = *(const uint4*)(p);
  f.q[1] = *(const uint4*)(p + 16);
  return f.v;
}
// B fragment (32x16 bf16, K x N) from row-major W[N][K]: lane<16 -> N=n0+lane, K in {k0..k0+7, k0+16..k0+23};
// lane>=16 -> N=n0+lane-16, K in {k0+8..k0+15, k0+24..k0+31}.
__device__ __forceinline__ v16bf load_b(const unsigned short* W, int ldw, int n0, int k0) {
  int lane = threadIdx.x & 31;
  const unsigned short* p = W + (size_t)(n0 + (lane & 15)) * ldw + k0 + ((lane >> 4) << 3);
  ABFrag f;
  f.q[0] = *(const uint4*)(p);
  f.q[1] = *(const uint4*)(p + 16);
  return f.v;
}

#define WMMA_BF16(a, b, c) \
  __builtin_amdgcn_wmma_f32_16x16x32_bf16(false, (a), false, (b), (short)0, (c), false, false)

// single 16x16 tile
__device__ __forceinline__ v8f gemm_loop(const unsigned short* A, int lda, int m0,
                                         const unsigned short* W, int ldw, int n0,
                                         int K, v8f acc) {
  for (int k = 0; k < K; k += 32)
    acc = WMMA_BF16(load_a(A, lda, m0, k), load_b(W, ldw, n0, k), acc);
  return acc;
}
// 1 M-tile x 2 N-tiles: share the A fragment
__device__ __forceinline__ void gemm_2n(const unsigned short* A, int lda, int m0,
                                        const unsigned short* W, int ldw, int n0a, int n0b,
                                        int K, v8f& c0, v8f& c1) {
  for (int k = 0; k < K; k += 32) {
    v16bf a  = load_a(A, lda, m0, k);
    v16bf b0 = load_b(W, ldw, n0a, k);
    v16bf b1 = load_b(W, ldw, n0b, k);
    c0 = WMMA_BF16(a, b0, c0);
    c1 = WMMA_BF16(a, b1, c1);
  }
}
// 2 M-tiles (m=0,16) x 2 N-tiles: 4 fragment loads feed 4 WMMAs per k-step
__device__ __forceinline__ void gemm_quad(const unsigned short* A, int lda,
                                          const unsigned short* W, int ldw,
                                          int n0a, int n0b, int K,
                                          v8f& c00, v8f& c01, v8f& c10, v8f& c11) {
  for (int k = 0; k < K; k += 32) {
    v16bf a0 = load_a(A, lda, 0,  k);
    v16bf a1 = load_a(A, lda, 16, k);
    v16bf b0 = load_b(W, ldw, n0a, k);
    v16bf b1 = load_b(W, ldw, n0b, k);
    c00 = WMMA_BF16(a0, b0, c00);
    c10 = WMMA_BF16(a1, b0, c10);
    c01 = WMMA_BF16(a0, b1, c01);
    c11 = WMMA_BF16(a1, b1, c11);
  }
}

// C/D layout: lane<16 -> N=n0+lane, M=m0+v ; lane>=16 -> N=n0+lane-16, M=m0+8+v
__device__ __forceinline__ void store_relu_bf16(unsigned short* D, int ldd, int m0, int n0, F8 acc) {
  int lane = threadIdx.x & 31;
  int n = n0 + (lane & 15);
  int mb = m0 + ((lane >> 4) << 3);
#pragma unroll
  for (int v = 0; v < 8; ++v) {
    float x = acc.f[v];
    D[(size_t)(mb + v) * ldd + n] = f2bf(x > 0.f ? x : 0.f);
  }
}
__device__ __forceinline__ void store_f32(float* D, int ldd, int m0, int n0, F8 acc) {
  int lane = threadIdx.x & 31;
  int n = n0 + (lane & 15);
  int mb = m0 + ((lane >> 4) << 3);
#pragma unroll
  for (int v = 0; v < 8; ++v) D[(size_t)(mb + v) * ldd + n] = acc.f[v];
}
__device__ __forceinline__ void init_acc(F8& a, float b) {
#pragma unroll
  for (int v = 0; v < 8; ++v) a.f[v] = b;
}

// ---------------------------------------------------------------------------
// Prep kernels
// ---------------------------------------------------------------------------
__global__ void k_convert2d(const float* __restrict__ src, unsigned short* __restrict__ dst,
                            int rows, int cols, int dstld, int coff) {
  int n = rows * cols;
  for (int i = blockIdx.x * blockDim.x + threadIdx.x; i < n; i += gridDim.x * blockDim.x) {
    int r = i / cols, c = i - r * cols;
    dst[(size_t)r * dstld + coff + c] = f2bf(src[i]);
  }
}
__global__ void k_combine_bias(const float* __restrict__ a, const float* __restrict__ b,
                               float* __restrict__ d, int n) {
  int i = blockIdx.x * blockDim.x + threadIdx.x;
  if (i < n) d[i] = a[i] + b[i];
}
__global__ void k_zero(float* __restrict__ p, int n) {
  for (int i = blockIdx.x * blockDim.x + threadIdx.x; i < n; i += gridDim.x * blockDim.x)
    p[i] = 0.f;
}
__global__ void k_initdec(const float* __restrict__ obs_traj, const float* __restrict__ obs_rel,
                          const float* __restrict__ embW, const float* __restrict__ embB,
                          unsigned short* __restrict__ xemb_bf, float* __restrict__ last_pos) {
  int row = blockIdx.x * blockDim.x + threadIdx.x;
  if (row >= NB) return;
  float rx = obs_rel[(OBS_LEN - 1) * NB * 2 + row * 2 + 0];
  float ry = obs_rel[(OBS_LEN - 1) * NB * 2 + row * 2 + 1];
  for (int e = 0; e < EMB; ++e)
    xemb_bf[(size_t)row * EMB + e] = f2bf(embW[e * 2] * rx + embW[e * 2 + 1] * ry + embB[e]);
  last_pos[row * 2 + 0] = obs_traj[(OBS_LEN - 1) * NB * 2 + row * 2 + 0];
  last_pos[row * 2 + 1] = obs_traj[(OBS_LEN - 1) * NB * 2 + row * 2 + 1];
}

// ---------------------------------------------------------------------------
// Encoder: embedding + 8-step LSTM (EH=64). 32 rows/WG, z = [emb|h] @ Wcat^T.
// ---------------------------------------------------------------------------
__global__ void __launch_bounds__(256)
k_encoder(const float* __restrict__ obs_rel,
          const float* __restrict__ embW, const float* __restrict__ embB,
          const unsigned short* __restrict__ Wcat, const float* __restrict__ bcomb,
          unsigned short* __restrict__ ctxin) {
  __shared__ unsigned short Xc[32 * 128];  // [emb(64) | h(64)] bf16
  __shared__ float Z[32 * 256];
  __shared__ float C[32 * 64];
  int tid = threadIdx.x, lane = tid & 31, wave = tid >> 5;
  int r0 = blockIdx.x * 32;
  for (int i = tid; i < 32 * 64; i += 256) {
    C[i] = 0.f;
    int j = i >> 6, u = i & 63;
    Xc[j * 128 + 64 + u] = 0;
  }
  __syncthreads();
  for (int t = 0; t < OBS_LEN; ++t) {
    for (int i = tid; i < 32 * 64; i += 256) {   // spatial embedding
      int j = i >> 6, e = i & 63;
      float rx = obs_rel[t * NB * 2 + (r0 + j) * 2 + 0];
      float ry = obs_rel[t * NB * 2 + (r0 + j) * 2 + 1];
      Xc[j * 128 + e] = f2bf(embW[e * 2] * rx + embW[e * 2 + 1] * ry + embB[e]);
    }
    __syncthreads();
    {                                            // z[32,256]: each wave 2 n-tiles x 2 m-tiles
      int n0a = (wave * 2 + 0) * 16, n0b = (wave * 2 + 1) * 16;
      F8 c00, c01, c10, c11;
      init_acc(c00, bcomb[n0a + (lane & 15)]); c10 = c00;
      init_acc(c01, bcomb[n0b + (lane & 15)]); c11 = c01;
      gemm_quad(Xc, 128, Wcat, 128, n0a, n0b, 128, c00.v, c01.v, c10.v, c11.v);
      store_f32(Z, 256, 0,  n0a, c00);
      store_f32(Z, 256, 16, n0a, c10);
      store_f32(Z, 256, 0,  n0b, c01);
      store_f32(Z, 256, 16, n0b, c11);
    }
    __syncthreads();
    for (int i = tid; i < 32 * 64; i += 256) {   // gates
      int j = i >> 6, u = i & 63;
      float zi = Z[j * 256 + u], zf = Z[j * 256 + 64 + u];
      float zg = Z[j * 256 + 128 + u], zo = Z[j * 256 + 192 + u];
      float cc = sigm(zf) * C[i] + sigm(zi) * tanhf(zg);
      C[i] = cc;
      float h = sigm(zo) * tanhf(cc);
      Xc[j * 128 + 64 + u] = f2bf(h);
      if (t == OBS_LEN - 1) ctxin[(size_t)(r0 + j) * 1088 + u] = f2bf(h);
    }
    __syncthreads();
  }
}

// ---------------------------------------------------------------------------
// Fused PoolHiddenNet: 1 WG per (g,i); rows = j. Max-over-j folded into H2 tiles.
// ---------------------------------------------------------------------------
__global__ void __launch_bounds__(256)
k_pool(const float* __restrict__ pos,
       const unsigned short* __restrict__ hbf, int hld, int Hdim,
       const float* __restrict__ seW, const float* __restrict__ seb,
       const unsigned short* __restrict__ W1, const float* __restrict__ b1,
       const unsigned short* __restrict__ W2, const float* __restrict__ b2,
       unsigned short* __restrict__ outbf, int outld, int outoff) {
  __shared__ unsigned short X[32 * 192];   // max K1 = 192
  __shared__ unsigned short H1[32 * 512];
  int tid = threadIdx.x, lane = tid & 31, wave = tid >> 5;
  int g = blockIdx.x >> 5, i = blockIdx.x & 31;
  int rb = g * 32;
  int K1 = 64 + Hdim;
  // warm L2/near caches for the weights (global_prefetch_b8)
  __builtin_prefetch(W1 + ((size_t)tid << 6), 0, 3);
  __builtin_prefetch(W2 + ((size_t)tid << 6), 0, 3);
  float pix = pos[(rb + i) * 2 + 0], piy = pos[(rb + i) * 2 + 1];
  for (int idx = tid; idx < 32 * 64; idx += 256) {          // rel embedding (linear)
    int j = idx >> 6, e = idx & 63;
    float rx = pos[(rb + j) * 2 + 0] - pix;
    float ry = pos[(rb + j) * 2 + 1] - piy;
    X[j * K1 + e] = f2bf(seW[e * 2] * rx + seW[e * 2 + 1] * ry + seb[e]);
  }
  for (int idx = tid; idx < 32 * Hdim; idx += 256) {        // partner hidden (bf16 copy)
    int j = idx / Hdim, u = idx - j * Hdim;
    X[j * K1 + 64 + u] = hbf[(size_t)(rb + j) * hld + u];
  }
  __syncthreads();
  for (int q = 0; q < 2; ++q) {                             // H1[32,512]: 4 n-tiles/wave
    int n0a = (wave * 4 + 2 * q + 0) * 16;
    int n0b = (wave * 4 + 2 * q + 1) * 16;
    F8 c00, c01, c10, c11;
    init_acc(c00, b1[n0a + (lane & 15)]); c10 = c00;
    init_acc(c01, b1[n0b + (lane & 15)]); c11 = c01;
    gemm_quad(X, K1, W1, K1, n0a, n0b, K1, c00.v, c01.v, c10.v, c11.v);
    store_relu_bf16(H1, 512, 0,  n0a, c00);
    store_relu_bf16(H1, 512, 16, n0a, c10);
    store_relu_bf16(H1, 512, 0,  n0b, c01);
    store_relu_bf16(H1, 512, 16, n0b, c11);
  }
  __syncthreads();
  for (int q = 0; q < 4; ++q) {                             // H2 + max over j: 8 n-tiles/wave
    int n0a = (wave * 8 + 2 * q + 0) * 16;
    int n0b = (wave * 8 + 2 * q + 1) * 16;
    F8 c00, c01, c10, c11;
    init_acc(c00, b2[n0a + (lane & 15)]); c10 = c00;
    init_acc(c01, b2[n0b + (lane & 15)]); c11 = c01;
    gemm_quad(H1, 512, W2, 512, n0a, n0b, 512, c00.v, c01.v, c10.v, c11.v);
    float ma = 0.f, mb2 = 0.f;                              // relu >= 0 so 0-init exact
#pragma unroll
    for (int v = 0; v < 8; ++v) {
      ma  = fmaxf(ma,  fmaxf(c00.f[v], c10.f[v]));
      mb2 = fmaxf(mb2, fmaxf(c01.f[v], c11.f[v]));
    }
    ma  = fmaxf(ma,  __shfl_xor(ma, 16));
    mb2 = fmaxf(mb2, __shfl_xor(mb2, 16));
    if (lane < 16) {
      outbf[(size_t)(rb + i) * outld + outoff + n0a + lane] = f2bf(ma);
      outbf[(size_t)(rb + i) * outld + outoff + n0b + lane] = f2bf(mb2);
    }
  }
}

// ---------------------------------------------------------------------------
// Fused 2-layer MLP (Linear->ReLU->Linear->ReLU), hidden 1024, out 128 (fp32).
// ---------------------------------------------------------------------------
__global__ void __launch_bounds__(256)
k_mlp2(const unsigned short* __restrict__ Xbf, int xld, int Kin,
       const unsigned short* __restrict__ W1, const float* __restrict__ b1,
       const unsigned short* __restrict__ W2, const float* __restrict__ b2,
       float* __restrict__ out, int outld) {
  __shared__ unsigned short H1[16 * 1024];
  int tid = threadIdx.x, lane = tid & 31, wave = tid >> 5;
  int r0 = blockIdx.x * 16;
  __builtin_prefetch(W1 + ((size_t)tid << 6), 0, 3);
  __builtin_prefetch(W2 + ((size_t)tid << 6), 0, 3);
  const unsigned short* A = Xbf + (size_t)r0 * xld;
  for (int q = 0; q < 4; ++q) {                             // H1[16,1024]: 8 n-tiles/wave
    int n0a = (wave * 8 + 2 * q + 0) * 16;
    int n0b = (wave * 8 + 2 * q + 1) * 16;
    F8 c0, c1;
    init_acc(c0, b1[n0a + (lane & 15)]);
    init_acc(c1, b1[n0b + (lane & 15)]);
    gemm_2n(A, xld, 0, W1, Kin, n0a, n0b, Kin, c0.v, c1.v);
    store_relu_bf16(H1, 1024, 0, n0a, c0);
    store_relu_bf16(H1, 1024, 0, n0b, c1);
  }
  __syncthreads();
  {                                                          // H2[16,128]: 1 n-tile/wave
    int n0 = wave * 16;
    F8 acc;
    init_acc(acc, b2[n0 + (lane & 15)]);
    acc.v = gemm_loop(H1, 1024, 0, W2, 1024, n0, 1024, acc.v);
    int n = n0 + (lane & 15);
    int mb = (lane >> 4) << 3;
#pragma unroll
    for (int v = 0; v < 8; ++v) {
      float x = acc.f[v];
      out[(size_t)(r0 + mb + v) * outld + n] = x > 0.f ? x : 0.f;
    }
  }
}

// ---------------------------------------------------------------------------
// Decoder LSTM step (DH=128) + pos linear + curr_pos + next input embedding.
// ---------------------------------------------------------------------------
__global__ void __launch_bounds__(256)
k_dec_lstm(const unsigned short* __restrict__ xemb_bf,          // [NB][64]
           float* __restrict__ h32, float* __restrict__ c32,    // [NB][128]
           const unsigned short* __restrict__ Wcat, const float* __restrict__ bcomb,
           const float* __restrict__ posW, const float* __restrict__ posb,
           const float* __restrict__ dembW, const float* __restrict__ dembB,
           float* __restrict__ last_pos, float* __restrict__ pos_cur,
           unsigned short* __restrict__ hbf,
           unsigned short* __restrict__ dmlpin,
           unsigned short* __restrict__ xemb_out,
           float* __restrict__ rel_out) {
  __shared__ unsigned short Xc[16 * 192];
  __shared__ float Z[16 * 512];
  __shared__ float HL[16 * 128];
  __shared__ float RP[16 * 2];
  int tid = threadIdx.x, lane = tid & 31, wave = tid >> 5;
  int r0 = blockIdx.x * 16;
  for (int idx = tid; idx < 16 * 64; idx += 256) {
    int j = idx >> 6, e = idx & 63;
    Xc[j * 192 + e] = xemb_bf[(size_t)(r0 + j) * 64 + e];
  }
  for (int idx = tid; idx < 16 * 128; idx += 256) {
    int j = idx >> 7, u = idx & 127;
    Xc[j * 192 + 64 + u] = f2bf(h32[(size_t)(r0 + j) * 128 + u]);
  }
  __syncthreads();
  for (int q = 0; q < 2; ++q) {                              // z[16,512]: 4 n-tiles/wave
    int n0a = (wave * 4 + 2 * q + 0) * 16;
    int n0b = (wave * 4 + 2 * q + 1) * 16;
    F8 c0, c1;
    init_acc(c0, bcomb[n0a + (lane & 15)]);
    init_acc(c1, bcomb[n0b + (lane & 15)]);
    gemm_2n(Xc, 192, 0, Wcat, 192, n0a, n0b, 192, c0.v, c1.v);
    store_f32(Z, 512, 0, n0a, c0);
    store_f32(Z, 512, 0, n0b, c1);
  }
  __syncthreads();
  for (int idx = tid; idx < 16 * 128; idx += 256) {          // gate math, update h/c
    int j = idx >> 7, u = idx & 127;
    int row = r0 + j;
    float zi = Z[j * 512 + u],       zf = Z[j * 512 + 128 + u];
    float zg = Z[j * 512 + 256 + u], zo = Z[j * 512 + 384 + u];
    float cc = sigm(zf) * c32[(size_t)row * 128 + u] + sigm(zi) * tanhf(zg);
    c32[(size_t)row * 128 + u] = cc;
    float h = sigm(zo) * tanhf(cc);
    h32[(size_t)row * 128 + u] = h;
    HL[j * 128 + u] = h;
    unsigned short hb = f2bf(h);
    hbf[(size_t)row * 128 + u] = hb;
    dmlpin[(size_t)row * 1152 + u] = hb;
  }
  __syncthreads();
  if (tid < 32) {                                            // rel_pos = h @ posW^T + posb
    int j = tid >> 1, comp = tid & 1;
    int row = r0 + j;
    float s = posb[comp];
    for (int k = 0; k < 128; ++k) s += posW[comp * 128 + k] * HL[j * 128 + k];
    float cp = s + last_pos[row * 2 + comp];
    rel_out[row * 2 + comp] = s;
    last_pos[row * 2 + comp] = cp;
    pos_cur[row * 2 + comp] = cp;
    RP[j * 2 + comp] = s;
  }
  __syncthreads();
  for (int idx = tid; idx < 16 * 64; idx += 256) {           // next input embedding
    int j = idx >> 6, e = idx & 63;
    float v = dembW[e * 2] * RP[j * 2 + 0] + dembW[e * 2 + 1] * RP[j * 2 + 1] + dembB[e];
    xemb_out[(size_t)(r0 + j) * 64 + e] = f2bf(v);
  }
}

// ---------------------------------------------------------------------------
// Host orchestration
// ---------------------------------------------------------------------------
enum {
  IN_OBS = 0, IN_OBSREL, IN_SSE,
  P_ENC_EMB_W, P_ENC_EMB_B, P_ENC_WIH, P_ENC_WHH, P_ENC_BIH, P_ENC_BHH,
  P_EP_SE_W, P_EP_SE_B, P_EP_W1, P_EP_B1, P_EP_W2, P_EP_B2,
  P_CTX_W1, P_CTX_B1, P_CTX_W2, P_CTX_B2,
  P_DEC_EMB_W, P_DEC_EMB_B, P_DEC_WIH, P_DEC_WHH, P_DEC_BIH, P_DEC_BHH,
  P_POS_W, P_POS_B,
  P_DP_SE_W, P_DP_SE_B, P_DP_W1, P_DP_B1, P_DP_W2, P_DP_B2,
  P_DMLP_W1, P_DMLP_B1, P_DMLP_W2, P_DMLP_B2
};

static inline int cvt_blocks(int n) {
  int b = (n + 255) / 256;
  return b > 2048 ? 2048 : b;
}

extern "C" void kernel_launch(void* const* d_in, const int* in_sizes, int n_in,
                              void* d_out, int out_size, void* d_ws, size_t ws_size,
                              hipStream_t stream) {
  (void)in_sizes; (void)n_in; (void)out_size; (void)ws_size;
  const float* F[40];
  for (int i = 0; i < 37; ++i) F[i] = (const float*)d_in[i];

  // -------- workspace carve-out (256B aligned) --------
  char* w = (char*)d_ws; size_t off = 0;
  auto alloc = [&](size_t bytes) -> void* {
    void* p = w + off; off = (off + bytes + 255) & ~(size_t)255; return p;
  };
  unsigned short* enc_cat  = (unsigned short*)alloc(256 * 128 * 2);
  float*          enc_bc   = (float*)alloc(256 * 4);
  unsigned short* epW1     = (unsigned short*)alloc((size_t)512 * 128 * 2);
  unsigned short* epW2     = (unsigned short*)alloc((size_t)1024 * 512 * 2);
  unsigned short* ctxW1    = (unsigned short*)alloc((size_t)1024 * 1088 * 2);
  unsigned short* ctxW2    = (unsigned short*)alloc((size_t)128 * 1024 * 2);
  unsigned short* dec_cat  = (unsigned short*)alloc((size_t)512 * 192 * 2);
  float*          dec_bc   = (float*)alloc(512 * 4);
  unsigned short* dpW1     = (unsigned short*)alloc((size_t)512 * 192 * 2);
  unsigned short* dpW2     = (unsigned short*)alloc((size_t)1024 * 512 * 2);
  unsigned short* dmW1     = (unsigned short*)alloc((size_t)1024 * 1152 * 2);
  unsigned short* dmW2     = (unsigned short*)alloc((size_t)128 * 1024 * 2);
  unsigned short* ctxin    = (unsigned short*)alloc((size_t)NB * 1088 * 2);
  unsigned short* dmlpin   = (unsigned short*)alloc((size_t)NB * 1152 * 2);
  float*          h32      = (float*)alloc((size_t)NB * 128 * 4);
  float*          c32      = (float*)alloc((size_t)NB * 128 * 4);
  unsigned short* hbf      = (unsigned short*)alloc((size_t)NB * 128 * 2);
  unsigned short* xemb     = (unsigned short*)alloc((size_t)NB * 64 * 2);
  float*          last_pos = (float*)alloc((size_t)NB * 2 * 4);
  float*          pos_cur  = (float*)alloc((size_t)NB * 2 * 4);

  // -------- weight prep (fp32 -> bf16, LSTM weights concatenated) --------
  auto cvt = [&](const float* src, unsigned short* dst, int rows, int cols, int ld, int coff) {
    k_convert2d<<<cvt_blocks(rows * cols), 256, 0, stream>>>(src, dst, rows, cols, ld, coff);
  };
  cvt(F[P_ENC_WIH], enc_cat, 256, 64, 128, 0);
  cvt(F[P_ENC_WHH], enc_cat, 256, 64, 128, 64);
  cvt(F[P_EP_W1],   epW1,    512, 128, 128, 0);
  cvt(F[P_EP_W2],   epW2,   1024, 512, 512, 0);
  cvt(F[P_CTX_W1],  ctxW1,  1024, 1088, 1088, 0);
  cvt(F[P_CTX_W2],  ctxW2,   128, 1024, 1024, 0);
  cvt(F[P_DEC_WIH], dec_cat, 512, 64, 192, 0);
  cvt(F[P_DEC_WHH], dec_cat, 512, 128, 192, 64);
  cvt(F[P_DP_W1],   dpW1,    512, 192, 192, 0);
  cvt(F[P_DP_W2],   dpW2,   1024, 512, 512, 0);
  cvt(F[P_DMLP_W1], dmW1,   1024, 1152, 1152, 0);
  cvt(F[P_DMLP_W2], dmW2,    128, 1024, 1024, 0);
  k_combine_bias<<<1, 256, 0, stream>>>(F[P_ENC_BIH], F[P_ENC_BHH], enc_bc, 256);
  k_combine_bias<<<2, 256, 0, stream>>>(F[P_DEC_BIH], F[P_DEC_BHH], dec_bc, 512);
  k_zero<<<256, 256, 0, stream>>>(c32, NB * 128);
  k_initdec<<<4, 256, 0, stream>>>(F[IN_OBS], F[IN_OBSREL], F[P_DEC_EMB_W], F[P_DEC_EMB_B],
                                   xemb, last_pos);

  // -------- encoder LSTM -> h_enc (bf16) into ctxin[:, :64] --------
  k_encoder<<<NB / 32, 256, 0, stream>>>(F[IN_OBSREL], F[P_ENC_EMB_W], F[P_ENC_EMB_B],
                                         enc_cat, enc_bc, ctxin);

  // -------- encoder-side pool: pos = obs_traj[-1]; h from ctxin[:, :64] --------
  const float* end_pos = F[IN_OBS] + (size_t)(OBS_LEN - 1) * NB * 2;
  k_pool<<<NB, 256, 0, stream>>>(end_pos, ctxin, 1088, 64,
                                 F[P_EP_SE_W], F[P_EP_SE_B],
                                 epW1, F[P_EP_B1], epW2, F[P_EP_B2],
                                 ctxin, 1088, 64);

  // -------- decoder context MLP -> initial hidden h32 --------
  k_mlp2<<<NB / 16, 256, 0, stream>>>(ctxin, 1088, 1088,
                                      ctxW1, F[P_CTX_B1], ctxW2, F[P_CTX_B2],
                                      h32, 128);

  // -------- decoder: 12 steps of (LSTM -> pool -> MLP) --------
  float* out = (float*)d_out;
  for (int t = 0; t < PRED_LEN; ++t) {
    k_dec_lstm<<<NB / 16, 256, 0, stream>>>(xemb, h32, c32, dec_cat, dec_bc,
                                            F[P_POS_W], F[P_POS_B],
                                            F[P_DEC_EMB_W], F[P_DEC_EMB_B],
                                            last_pos, pos_cur, hbf, dmlpin, xemb,
                                            out + (size_t)t * NB * 2);
    k_pool<<<NB, 256, 0, stream>>>(pos_cur, hbf, 128, 128,
                                   F[P_DP_SE_W], F[P_DP_SE_B],
                                   dpW1, F[P_DP_B1], dpW2, F[P_DP_B2],
                                   dmlpin, 1152, 128);
    k_mlp2<<<NB / 16, 256, 0, stream>>>(dmlpin, 1152, 1152,
                                        dmW1, F[P_DMLP_B1], dmW2, F[P_DMLP_B2],
                                        h32, 128);
  }
}
